// MultiHeadAttention_8323646620393
// MI455X (gfx1250) — matrix operand order repaired and checked by reading, not compiled, not run on hardware
//
#include <hip/hip_runtime.h>
#include <hip/hip_bf16.h>

// ---------------------------------------------------------------------------
// MHA forward for MI455X (gfx1250, wave32, WMMA).
// Compute-bound (~69 GFLOP vs ~100MB HBM traffic @23.3TB/s) -> bf16 WMMA:
//   v_wmma_f32_16x16x32_bf16, f32 accumulate, softmax in f32.
// Data movement: double-buffered GLOBAL_LOAD_ASYNC_TO_LDS_B128 (ASYNCcnt)
// so tile copies overlap WMMA work; row-sums via a P @ ones WMMA.
// ---------------------------------------------------------------------------

typedef __bf16        v16bf __attribute__((ext_vector_type(16)));
typedef float         v8f   __attribute__((ext_vector_type(8)));
typedef unsigned int  v4u   __attribute__((ext_vector_type(4)));
typedef int           gv4i  __attribute__((vector_size(16)));    // builtin's pointee type

typedef __attribute__((address_space(1))) gv4i* gptr_t;
typedef __attribute__((address_space(3))) gv4i* lptr_t;

#define EMBED 1024
#define SEQ   2048
#define HEADS 16
#define HDIM  64
#define BATCH 2
#define MTOT  (BATCH * SEQ)   // 4096 rows in all GEMMs

// 32-byte fragment: two 16B chunks, viewed as 16 bf16 for the WMMA builtin.
union FragU {
    v4u   q[2];
    v16bf v;
};

static __device__ __forceinline__ v8f wmma_bf16(const FragU& a, const FragU& b, v8f c) {
    return __builtin_amdgcn_wmma_f32_16x16x32_bf16(false, a.v, false, b.v, (short)0, c,
                                                   false, false);
}

static __device__ __forceinline__ unsigned short f2bf(float f) {
    union { float f; unsigned u; } v; v.f = f;
    unsigned r = v.u + 0x7FFFu + ((v.u >> 16) & 1u);   // round-to-nearest-even
    return (unsigned short)(r >> 16);
}

// 16B async copy global -> LDS (ASYNCcnt-tracked, per-lane addresses).
static __device__ __forceinline__ void async_cp16(const v4u* g, v4u* l) {
    __builtin_amdgcn_global_load_async_to_lds_b128((gptr_t)g, (lptr_t)l, 0, 0);
}

// ---------------------------------------------------------------------------
// fp32 -> bf16 conversion (grid-stride)
// ---------------------------------------------------------------------------
__global__ void cvt_f32_bf16(const float* __restrict__ src,
                             unsigned short* __restrict__ dst, int n) {
    for (int i = blockIdx.x * blockDim.x + threadIdx.x; i < n;
         i += gridDim.x * blockDim.x)
        dst[i] = f2bf(src[i]);
}

// ---------------------------------------------------------------------------
// bf16 GEMM:  C[m,n] = sum_k A[m,k] * W[n,k] + bias[n]        (torch Linear)
// A: [MTOT,1024] bf16 row-major.  W: [1024,1024] bf16 row-major (N,K).
// Block tile 128x128x32, 8 waves (2x4), wave tile 64x32 (4x2 WMMA frags).
// K tiles staged via double-buffered async-to-LDS copies.
// MODE 0: bf16 out [B,H,S,D] * scale    (Q with 1/sqrt(D) folded in, K)
// MODE 1: bf16 out [B,H,D,S]            (V transposed for PV fragment loads)
// MODE 2: f32  out [MTOT,EMBED]         (final projection -> d_out)
// ---------------------------------------------------------------------------
template <int MODE>
__global__ __launch_bounds__(256)
void gemm_wmma_bf16(const unsigned short* __restrict__ A,
                    const unsigned short* __restrict__ W,
                    const float* __restrict__ bias,
                    void* __restrict__ out, float scale) {
    __shared__ v4u AsV[2][512];   // 128 rows x 32 bf16 per buffer
    __shared__ v4u BsV[2][512];

    const int tid  = threadIdx.x;
    const int lane = tid & 31;
    const int wave = tid >> 5;
    const int wm   = wave >> 2;         // 0..1 (M)
    const int wn   = wave & 3;          // 0..3 (N)
    const int half = lane >> 4;         // 0/1  (16-lane half)
    const int l16  = lane & 15;
    const int mBase = blockIdx.x * 128;
    const int nBase = blockIdx.y * 128;

    const v4u* Ag = reinterpret_cast<const v4u*>(A);   // row stride 128 v4u
    const v4u* Wg = reinterpret_cast<const v4u*>(W);

    // 512 v4u per tile per matrix; 256 threads x 2 chunks each, async.
    auto stage = [&](int k0, int buf) {
        const int kq = k0 >> 3;
#pragma unroll
        for (int t = 0; t < 2; ++t) {
            int i = tid + t * 256;
            int row = i >> 2, ch = i & 3;
            async_cp16(&Ag[(size_t)(mBase + row) * 128 + kq + ch], &AsV[buf][i]);
            async_cp16(&Wg[(size_t)(nBase + row) * 128 + kq + ch], &BsV[buf][i]);
        }
    };

    v8f acc[4][2];
#pragma unroll
    for (int i = 0; i < 4; ++i)
#pragma unroll
        for (int j = 0; j < 2; ++j) acc[i][j] = (v8f)0.0f;

    stage(0, 0);
    for (int k0 = 0; k0 < EMBED; k0 += 32) {
        const int buf = (k0 >> 5) & 1;
        if (k0 + 32 < EMBED) {
            stage(k0 + 32, buf ^ 1);                    // overlap next copy
            asm volatile("s_wait_asynccnt 0x4" ::: "memory");  // this tile done
        } else {
            asm volatile("s_wait_asynccnt 0x0" ::: "memory");
        }
        __syncthreads();

        // A frag (16x32): lane<16 -> K chunks {0..7,16..23}; lane>=16 -> {8..15,24..31}
        FragU aF[4], bF[2];
#pragma unroll
        for (int i = 0; i < 4; ++i) {
            int row = wm * 64 + i * 16 + l16;
            aF[i].q[0] = AsV[buf][row * 4 + half];
            aF[i].q[1] = AsV[buf][row * 4 + 2 + half];
        }
        // B frag (32x16): lane holds col N=l16, K chunks {0..7,16..23} / {8..15,24..31}
#pragma unroll
        for (int j = 0; j < 2; ++j) {
            int col = wn * 32 + j * 16 + l16;
            bF[j].q[0] = BsV[buf][col * 4 + half];
            bF[j].q[1] = BsV[buf][col * 4 + 2 + half];
        }
#pragma unroll
        for (int i = 0; i < 4; ++i)
#pragma unroll
            for (int j = 0; j < 2; ++j)
                acc[i][j] = wmma_bf16(aF[i], bF[j], acc[i][j]);
        __syncthreads();                                // protect buf reuse
    }

    // Epilogue. D frag: lane L, vgpr r -> M = r + 8*(L/16), N = L%16.
#pragma unroll
    for (int j = 0; j < 2; ++j) {
        const int nCol = nBase + wn * 32 + j * 16 + l16;
        const float bj = bias[nCol];
#pragma unroll
        for (int i = 0; i < 4; ++i) {
#pragma unroll
            for (int r = 0; r < 8; ++r) {
                const int   mRow = mBase + wm * 64 + i * 16 + r + 8 * half;
                const float v    = (acc[i][j][r] + bj) * scale;
                if (MODE == 0) {
                    int b = mRow >> 11, s = mRow & (SEQ - 1);
                    int h = nCol >> 6,  d = nCol & (HDIM - 1);
                    ((unsigned short*)out)[(((size_t)(b * HEADS + h) * SEQ + s) * HDIM + d)] = f2bf(v);
                } else if (MODE == 1) {
                    int b = mRow >> 11, s = mRow & (SEQ - 1);
                    int h = nCol >> 6,  d = nCol & (HDIM - 1);
                    ((unsigned short*)out)[(((size_t)(b * HEADS + h) * HDIM + d) * SEQ + s)] = f2bf(v);
                } else {
                    ((float*)out)[(size_t)mRow * EMBED + nCol] = v;
                }
            }
        }
    }
}

// ---------------------------------------------------------------------------
// Flash-style attention. Q pre-scaled by 1/sqrt(D).
// Q,K: [B,H,S,D] bf16.  Vt: [B,H,D,S] bf16.  O: [B,S,E] bf16.
// Block = 8 waves sharing one (b,h) and 128 q rows; K/V tiles (32 keys) are
// staged once per block into LDS with double-buffered async copies.
// Per tile: 4 score WMMAs, online softmax (max via shfl in 16-lane halves,
// row-sum via P @ ones WMMA), P through per-wave LDS, 4 PV WMMAs.
// ---------------------------------------------------------------------------
__global__ __launch_bounds__(256)
void attention_wmma(const unsigned short* __restrict__ Q,
                    const unsigned short* __restrict__ K,
                    const unsigned short* __restrict__ Vt,
                    unsigned short* __restrict__ O) {
    __shared__ v4u Ks[2][32 * 8];                // 32 keys x 64 d (bf16)
    __shared__ v4u Vs[2][64 * 4];                // 64 d rows x 32 s (bf16, Vt)
    __shared__ v4u PldsV[8][64];                 // 16x32 bf16 per wave

    const int tid  = threadIdx.x;
    const int lane = tid & 31;
    const int wave = tid >> 5;
    const int half = lane >> 4;
    const int l16  = lane & 15;
    const int bh   = blockIdx.y;                 // b*HEADS + h
    const int b    = bh >> 4, h = bh & 15;
    const int q0   = blockIdx.x * 128 + wave * 16;

    const v4u* Qg = reinterpret_cast<const v4u*>(Q + (size_t)bh * SEQ * HDIM);  // row: 8 v4u
    const v4u* Kg = reinterpret_cast<const v4u*>(K + (size_t)bh * SEQ * HDIM);
    const v4u* Vg = reinterpret_cast<const v4u*>(Vt + (size_t)bh * HDIM * SEQ); // row: 256 v4u

    // Cooperative async staging of one 32-key K/V tile (2 asyncs / thread).
    auto stageKV = [&](int j, int buf) {
        const int kb = j * 32;
        {   int row = tid >> 3, ch = tid & 7;    // K: 32 rows x 8 chunks
            async_cp16(&Kg[(size_t)(kb + row) * 8 + ch], &Ks[buf][row * 8 + ch]); }
        {   int row = tid >> 2, ch = tid & 3;    // Vt: 64 rows x 4 chunks
            async_cp16(&Vg[(size_t)row * 256 + (kb >> 3) + ch], &Vs[buf][row * 4 + ch]); }
    };

    // Q as two A-fragments (K-dim 0..31 and 32..63), loaded once.
    FragU qa[2];
#pragma unroll
    for (int c = 0; c < 2; ++c) {
        int row = q0 + l16;
        qa[c].q[0] = Qg[(size_t)row * 8 + c * 4 + half];
        qa[c].q[1] = Qg[(size_t)row * 8 + c * 4 + 2 + half];
    }

    // all-ones B fragment for row-sum WMMA (bf16 1.0 = 0x3F80)
    FragU onesB;
    onesB.q[0] = (v4u)0x3F803F80u;
    onesB.q[1] = (v4u)0x3F803F80u;

    v8f o[4];
#pragma unroll
    for (int t = 0; t < 4; ++t) o[t] = (v8f)0.0f;
    float mstat[8], lstat[8], corr[8];
#pragma unroll
    for (int r = 0; r < 8; ++r) { mstat[r] = -__builtin_inff(); lstat[r] = 0.0f; }

    unsigned short* Pw = reinterpret_cast<unsigned short*>(&PldsV[wave][0]);

    stageKV(0, 0);
    for (int j = 0; j < SEQ / 32; ++j) {
        const int buf = j & 1;
        if (j + 1 < SEQ / 32) {
            stageKV(j + 1, buf ^ 1);
            asm volatile("s_wait_asynccnt 0x2" ::: "memory");
        } else {
            asm volatile("s_wait_asynccnt 0x0" ::: "memory");
        }
        __syncthreads();

        // ---- scores: S = Qhat @ K^T, 2 key sub-tiles of 16 ----------------
        v8f sc[2];
#pragma unroll
        for (int t = 0; t < 2; ++t) {
            sc[t] = (v8f)0.0f;
#pragma unroll
            for (int c = 0; c < 2; ++c) {       // kdim chunks 0..31 / 32..63
                FragU kf;
                int keyRow = t * 16 + l16;
                kf.q[0] = Ks[buf][keyRow * 8 + c * 4 + half];
                kf.q[1] = Ks[buf][keyRow * 8 + c * 4 + 2 + half];
                sc[t] = wmma_bf16(qa[c], kf, sc[t]);
            }
        }

        // ---- online softmax: max via shfl within 16-lane halves -----------
#pragma unroll
        for (int r = 0; r < 8; ++r) {
            float v0 = sc[0][r], v1 = sc[1][r];
            float tmax = fmaxf(v0, v1);
#pragma unroll
            for (int off = 1; off < 16; off <<= 1)
                tmax = fmaxf(tmax, __shfl_xor(tmax, off));
            float mnew = fmaxf(mstat[r], tmax);
            corr[r] = __expf(mstat[r] - mnew);
            mstat[r] = mnew;
            float p0 = __expf(v0 - mnew);
            float p1 = __expf(v1 - mnew);
#pragma unroll
            for (int t = 0; t < 4; ++t) o[t][r] *= corr[r];
            int row = r + 8 * half;
            Pw[row * 32 + l16]      = f2bf(p0);
            Pw[row * 32 + 16 + l16] = f2bf(p1);
        }
        // same-wave LDS RAW (DS ops in-order; wait + fence codegen)
        asm volatile("s_wait_dscnt 0x0" ::: "memory");

        // ---- P (16x32) as A fragment ------------------------------------
        FragU pF;
        pF.q[0] = PldsV[wave][l16 * 4 + half];
        pF.q[1] = PldsV[wave][l16 * 4 + 2 + half];

        // row sums on the matrix unit: D[m][n] = sum_k P[m][k]
        v8f sums = wmma_bf16(pF, onesB, (v8f)0.0f);
#pragma unroll
        for (int r = 0; r < 8; ++r)
            lstat[r] = lstat[r] * corr[r] + sums[r];

        // ---- PV: V (32x64) as 4 B fragments ------------------------------
#pragma unroll
        for (int t = 0; t < 4; ++t) {
            FragU vf;
            int dcol = t * 16 + l16;
            vf.q[0] = Vs[buf][dcol * 4 + half];
            vf.q[1] = Vs[buf][dcol * 4 + 2 + half];
            o[t] = wmma_bf16(pF, vf, o[t]);
        }
        __syncthreads();                        // protect buf reuse
    }

    // ---- finalize: O / l, write [B,S,E] bf16 ------------------------------
#pragma unroll
    for (int r = 0; r < 8; ++r) {
        float inv = 1.0f / lstat[r];
        int   s   = q0 + r + 8 * half;
#pragma unroll
        for (int t = 0; t < 4; ++t) {
            float v = o[t][r] * inv;
            O[(size_t)(b * SEQ + s) * EMBED + h * HDIM + t * 16 + l16] = f2bf(v);
        }
    }
}

// ---------------------------------------------------------------------------
extern "C" void kernel_launch(void* const* d_in, const int* in_sizes, int n_in,
                              void* d_out, int out_size, void* d_ws, size_t ws_size,
                              hipStream_t stream) {
    (void)in_sizes; (void)n_in; (void)out_size; (void)ws_size;
    const float* x  = (const float*)d_in[0];
    const float* Wq = (const float*)d_in[1];
    const float* bq = (const float*)d_in[2];
    const float* Wk = (const float*)d_in[3];
    const float* bk = (const float*)d_in[4];
    const float* Wv = (const float*)d_in[5];
    const float* bv = (const float*)d_in[6];
    const float* Wo = (const float*)d_in[7];
    const float* bo = (const float*)d_in[8];

    // Workspace layout (bf16 = ushort), ~48 MB total.
    unsigned short* xb  = (unsigned short*)d_ws;                 // 4096x1024
    unsigned short* wqb = xb  + (size_t)MTOT * EMBED;            // 1024x1024 each
    unsigned short* wkb = wqb + (size_t)EMBED * EMBED;
    unsigned short* wvb = wkb + (size_t)EMBED * EMBED;
    unsigned short* wob = wvb + (size_t)EMBED * EMBED;
    unsigned short* Qb  = wob + (size_t)EMBED * EMBED;           // [B,H,S,D]
    unsigned short* Kb  = Qb  + (size_t)MTOT * EMBED;
    unsigned short* Vtb = Kb  + (size_t)MTOT * EMBED;            // [B,H,D,S]
    unsigned short* Ab  = Vtb + (size_t)MTOT * EMBED;            // attn out [B,S,E]

    cvt_f32_bf16<<<512, 256, 0, stream>>>(x,  xb,  MTOT * EMBED);
    cvt_f32_bf16<<<256, 256, 0, stream>>>(Wq, wqb, EMBED * EMBED);
    cvt_f32_bf16<<<256, 256, 0, stream>>>(Wk, wkb, EMBED * EMBED);
    cvt_f32_bf16<<<256, 256, 0, stream>>>(Wv, wvb, EMBED * EMBED);
    cvt_f32_bf16<<<256, 256, 0, stream>>>(Wo, wob, EMBED * EMBED);

    dim3 gProj(MTOT / 128, EMBED / 128);            // 32 x 8
    gemm_wmma_bf16<0><<<gProj, 256, 0, stream>>>(xb, wqb, bq, Qb, 0.125f); // 1/sqrt(64)
    gemm_wmma_bf16<0><<<gProj, 256, 0, stream>>>(xb, wkb, bk, Kb, 1.0f);
    gemm_wmma_bf16<1><<<gProj, 256, 0, stream>>>(xb, wvb, bv, Vtb, 1.0f);

    dim3 gAttn(SEQ / 128, BATCH * HEADS);           // 16 x 32
    attention_wmma<<<gAttn, 256, 0, stream>>>(Qb, Kb, Vtb, Ab);

    gemm_wmma_bf16<2><<<gProj, 256, 0, stream>>>(Ab, wob, bo, d_out, 1.0f);
}
